// Attention_1477468750130
// MI455X (gfx1250) — matrix operand order repaired and checked by reading, not compiled, not run on hardware
//
#include <hip/hip_runtime.h>
#include <hip/hip_bf16.h>

// ---------------------------------------------------------------------------
// Types for WMMA (CDNA5 / gfx1250, wave32)
// ---------------------------------------------------------------------------
typedef __attribute__((ext_vector_type(16))) __bf16 v16bf;
typedef __attribute__((ext_vector_type(8)))  float  v8f;

#define T_LEN 2048
#define C_DIM 2048
#define N_HEADS 16
#define N_KV 4
#define HD 128

static __device__ __forceinline__ __bf16 us2bf(unsigned short u) {
    union { unsigned short u; __bf16 b; } c; c.u = u; return c.b;
}
static __device__ __forceinline__ unsigned short f2us(float f) {
    union { unsigned short u; __bf16 b; } c; c.b = (__bf16)f; return c.u;
}
static __device__ __forceinline__ float bf2f(unsigned short u) {
    return (float)us2bf(u);
}

// A-fragment (16x32 bf16, M x K): lane holds row M = m0 + (lane&15).
// lanes 0-15: K = 0..15; lanes 16-31: K = 16..31.
static __device__ __forceinline__ v16bf load_a_frag(const unsigned short* __restrict__ rowk0,
                                                    int lane) {
    const int kh = (lane & 16) ? 16 : 0;
    v16bf a;
#pragma unroll
    for (int i = 0; i < 8; ++i) a[i] = us2bf(rowk0[kh + i]);
#pragma unroll
    for (int i = 0; i < 8; ++i) a[8 + i] = us2bf(rowk0[8 + kh + i]);
    return a;
}

// B-fragment (32x16 bf16, K x N) from BT[N][K] (transposed) storage:
// lane holds col N = n0 + (lane&15); lanes 0-15: K=0..15, lanes 16-31: K=16..31.
static __device__ __forceinline__ v16bf load_b_frag(const unsigned short* __restrict__ colk0,
                                                    int lane) {
    const int ko = (lane & 16) ? 16 : 0;
    v16bf b;
#pragma unroll
    for (int i = 0; i < 16; ++i) b[i] = us2bf(colk0[ko + i]);
    return b;
}

// ---------------------------------------------------------------------------
// Conversion kernels
// ---------------------------------------------------------------------------
__global__ void cvt_f32_bf16(const float* __restrict__ src, unsigned short* __restrict__ dst,
                             int n) {
    int i = blockIdx.x * blockDim.x + threadIdx.x;
    if (i < n) dst[i] = f2us(src[i]);
}

// dst[c][r] = bf16(src[r][c]);  src: rows x cols f32, dst: cols x rows bf16
__global__ void transpose_cvt_f32_bf16(const float* __restrict__ src,
                                       unsigned short* __restrict__ dst,
                                       int rows, int cols) {
    int i = blockIdx.x * blockDim.x + threadIdx.x;
    if (i >= rows * cols) return;
    int r = i / cols, c = i - r * cols;
    dst[(size_t)c * rows + r] = f2us(src[i]);
}

__global__ void transpose_bf16(const unsigned short* __restrict__ src,
                               unsigned short* __restrict__ dst,
                               int rows, int cols) {
    int i = blockIdx.x * blockDim.x + threadIdx.x;
    if (i >= rows * cols) return;
    int r = i / cols, c = i - r * cols;
    dst[(size_t)c * rows + r] = src[i];
}

// ---------------------------------------------------------------------------
// RoPE (in place, bf16 buffer [T][nheads*128]); thread owns the (j, j+64) pair
// ---------------------------------------------------------------------------
__global__ void rope_kernel(unsigned short* __restrict__ buf, int nheads, int rowstride) {
    int idx = blockIdx.x * blockDim.x + threadIdx.x;
    int total = T_LEN * nheads * 64;
    if (idx >= total) return;
    int j = idx & 63;
    int t = idx >> 6;
    int h = t % nheads;
    int s = t / nheads;
    unsigned short* p = buf + (size_t)s * rowstride + h * HD;
    float q1 = bf2f(p[j]);
    float q2 = bf2f(p[j + 64]);
    float inv_ts = __powf(10000.0f, -(float)j * (1.0f / 64.0f));
    float ang = (float)s * inv_ts;
    float sn, cs;
    __sincosf(ang, &sn, &cs);
    p[j]      = f2us(q1 * cs - q2 * sn);
    p[j + 64] = f2us(q2 * cs + q1 * sn);
}

// ---------------------------------------------------------------------------
// WMMA GEMM: C[M][N] = A[M][K] * BT[N][K]^T    (bf16 in, f32 accum)
// block = 256 thr = 8 waves; block tile 128(M) x 64(N); wave tile 16 x 64.
// Shared B tile (64 cols x 32 k) is DMA'd to LDS with
// global_load_async_to_lds_b128 (ASYNCcnt), double-buffered so the next
// tile's DMA overlaps the current tile's WMMAs. A rows are wave-private and
// stay as direct (b128-vectorized) global loads.
// ---------------------------------------------------------------------------
template <int OUT_F32>
__global__ __launch_bounds__(256) void gemm_bf16_kernel(
    const unsigned short* __restrict__ A, const unsigned short* __restrict__ BT,
    void* __restrict__ Cout, int M, int N, int K) {
    __shared__ __align__(16) unsigned short btile[2][64 * 32];  // 2 x 4KB

    const int tid = threadIdx.x;
    const int lane = tid & 31;
    const int wave = tid >> 5;
    const int m0 = blockIdx.y * 128 + wave * 16;
    const int n0 = blockIdx.x * 64;
    const int colL = lane & 15;

    // Per-thread staging slice: 16 bytes of the 64x32 B tile.
    const int scol = tid >> 2;                 // 0..63 : local column
    const int sseg = (tid & 3) * 8;            // 0,8,16,24 : k element offset
    const unsigned short* sgsrc = BT + (size_t)(n0 + scol) * K + sseg;

    auto stage = [&](int buf, int k0) {
        // Low 32 bits of the generic pointer == wave-relative LDS byte address.
        unsigned lds = (unsigned)(unsigned long long)(&btile[buf][scol * 32 + sseg]);
        unsigned long long g = (unsigned long long)(sgsrc + k0);
        asm volatile("global_load_async_to_lds_b128 %0, %1, off"
                     :: "v"(lds), "v"(g) : "memory");
    };

    v8f zero8 = {0,0,0,0,0,0,0,0};
    v8f acc[4] = {zero8, zero8, zero8, zero8};

    const unsigned short* Arow = A + (size_t)(m0 + colL) * K;
    const int nk = K / 32;

    stage(0, 0);
    for (int kt = 0; kt < nk; ++kt) {
        const int buf = kt & 1;
        asm volatile("s_wait_asynccnt 0x0" ::: "memory");  // this wave's DMA slice done
        __syncthreads();                                   // whole tile visible
        if (kt + 1 < nk) stage(buf ^ 1, (kt + 1) * 32);    // overlap next DMA

        __builtin_prefetch(Arow + kt * 32 + 256, 0, 0);
        v16bf a = load_a_frag(Arow + kt * 32, lane);

        const int ko = (lane & 16) ? 16 : 0;
#pragma unroll
        for (int nt = 0; nt < 4; ++nt) {
            const unsigned short* bp = &btile[buf][(nt * 16 + colL) * 32 + ko];
            v16bf b;
#pragma unroll
            for (int i = 0; i < 16; ++i) b[i] = us2bf(bp[i]);
            acc[nt] = __builtin_amdgcn_wmma_f32_16x16x32_bf16(
                false, a, false, b, (short)0, acc[nt], false, false);
        }
        __syncthreads();  // all waves done reading buf before it is re-staged
    }

    const int rb = (lane & 16) ? 8 : 0;
#pragma unroll
    for (int nt = 0; nt < 4; ++nt) {
        const int col = n0 + nt * 16 + colL;
#pragma unroll
        for (int r = 0; r < 8; ++r) {
            const int row = m0 + rb + r;
            if (OUT_F32)
                ((float*)Cout)[(size_t)row * N + col] = acc[nt][r];
            else
                ((unsigned short*)Cout)[(size_t)row * N + col] = f2us(acc[nt][r]);
        }
    }
}

// ---------------------------------------------------------------------------
// Flash attention with tanh soft-cap + causal mask.
// grid = (16 q-groups, 16 heads); block = 256 thr = 8 waves;
// each wave owns 16 query rows and streams 32-key blocks.
// Qb [T][2048], Kb [T][512], VT [512][T], Ob [T][2048]  (all bf16)
// ---------------------------------------------------------------------------
__global__ __launch_bounds__(256) void flash_attn_kernel(
    const unsigned short* __restrict__ Qb, const unsigned short* __restrict__ Kb,
    const unsigned short* __restrict__ VT, unsigned short* __restrict__ Ob) {
    __shared__ unsigned short pbuf[8][16 * 32];  // per-wave P staging (C->A relayout)

    const int lane = threadIdx.x & 31;
    const int wave = threadIdx.x >> 5;
    const int hq = blockIdx.y;          // query head 0..15
    const int kv = hq & 3;              // kv head = hq % 4  (r = hq/4, h = hq%4)
    const int q0 = (blockIdx.x * 8 + wave) * 16;
    const int colL = lane & 15;
    const int rb = (lane & 16) ? 8 : 0;

    const float scale = 0.08838834764831845f;  // 1/sqrt(128)
    const float icap = 1.0f / 50.0f;

    // Load Q fragments for this wave's 16 rows (K = 128 -> 4 chunks of 32)
    v16bf qa[4];
    {
        const int s = q0 + colL;
        const unsigned short* qrow = Qb + (size_t)s * 2048 + hq * HD;
#pragma unroll
        for (int kc = 0; kc < 4; ++kc) qa[kc] = load_a_frag(qrow + kc * 32, lane);
    }

    v8f zero8 = {0,0,0,0,0,0,0,0};
    v8f oacc[8] = {zero8, zero8, zero8, zero8, zero8, zero8, zero8, zero8};
    float mrow[8], lrow[8];
#pragma unroll
    for (int r = 0; r < 8; ++r) { mrow[r] = -1e30f; lrow[r] = 0.0f; }

    const int kend = q0 + 16;  // causal: keys 0 .. q0+15
    for (int j0 = 0; j0 < kend; j0 += 32) {
        // ---- S = Q * K^T (16 x 32), two 16x16 tiles ----
        v8f sc0 = zero8, sc1 = zero8;
        {
            const int d0 = j0 + colL;
            const int d1 = j0 + 16 + colL;
            const unsigned short* kr0 = Kb + (size_t)d0 * 512 + kv * HD;
            const unsigned short* kr1 = Kb + (size_t)d1 * 512 + kv * HD;
#pragma unroll
            for (int kc = 0; kc < 4; ++kc) {
                v16bf b0 = load_b_frag(kr0 + kc * 32, lane);
                sc0 = __builtin_amdgcn_wmma_f32_16x16x32_bf16(
                    false, qa[kc], false, b0, (short)0, sc0, false, false);
                v16bf b1 = load_b_frag(kr1 + kc * 32, lane);
                sc1 = __builtin_amdgcn_wmma_f32_16x16x32_bf16(
                    false, qa[kc], false, b1, (short)0, sc1, false, false);
            }
        }

        // ---- scale, tanh soft-cap, causal mask ----
        float lt0[8], lt1[8];
#pragma unroll
        for (int r = 0; r < 8; ++r) {
            const int qrow = q0 + rb + r;
            float x0 = tanhf(sc0[r] * scale * icap) * 50.0f;
            float x1 = tanhf(sc1[r] * scale * icap) * 50.0f;
            lt0[r] = ((j0 + colL) <= qrow) ? x0 : -1e30f;
            lt1[r] = ((j0 + 16 + colL) <= qrow) ? x1 : -1e30f;
        }

        // ---- online softmax (row reductions within the 16-lane groups) ----
        float corr[8], p0[8], p1[8];
#pragma unroll
        for (int r = 0; r < 8; ++r) {
            float rm = fmaxf(lt0[r], lt1[r]);
#pragma unroll
            for (int off = 1; off < 16; off <<= 1)
                rm = fmaxf(rm, __shfl_xor(rm, off, 32));
            float mn = fmaxf(mrow[r], rm);
            corr[r] = __expf(mrow[r] - mn);
            mrow[r] = mn;
            p0[r] = __expf(lt0[r] - mn);
            p1[r] = __expf(lt1[r] - mn);
            float rs = p0[r] + p1[r];
#pragma unroll
            for (int off = 1; off < 16; off <<= 1)
                rs += __shfl_xor(rs, off, 32);
            lrow[r] = lrow[r] * corr[r] + rs;
        }
#pragma unroll
        for (int vt = 0; vt < 8; ++vt)
#pragma unroll
            for (int r = 0; r < 8; ++r) oacc[vt][r] *= corr[r];

        // ---- re-layout P: C-layout regs -> LDS -> A-layout fragment ----
#pragma unroll
        for (int r = 0; r < 8; ++r) {
            const int row = rb + r;
            pbuf[wave][row * 32 + colL]      = f2us(p0[r]);
            pbuf[wave][row * 32 + 16 + colL] = f2us(p1[r]);
        }
        asm volatile("s_wait_dscnt 0" ::: "memory");  // lanes read each other's stores
        v16bf pa;
        {
            const int kh = (lane & 16) ? 16 : 0;
            const unsigned short* pb = &pbuf[wave][colL * 32];
#pragma unroll
            for (int i = 0; i < 8; ++i) pa[i] = us2bf(pb[kh + i]);
#pragma unroll
            for (int i = 0; i < 8; ++i) pa[8 + i] = us2bf(pb[8 + kh + i]);
        }

        // ---- O += P * V  (V^T gives contiguous B fragments) ----
#pragma unroll
        for (int vt = 0; vt < 8; ++vt) {
            const int f = kv * HD + vt * 16 + colL;
            const unsigned short* vrow = VT + (size_t)f * T_LEN + j0;
            v16bf vb = load_b_frag(vrow, lane);
            oacc[vt] = __builtin_amdgcn_wmma_f32_16x16x32_bf16(
                false, pa, false, vb, (short)0, oacc[vt], false, false);
        }
    }

    // ---- normalize and store O[s][hq*128 + v] ----
#pragma unroll
    for (int vt = 0; vt < 8; ++vt) {
        const int f = hq * HD + vt * 16 + colL;
#pragma unroll
        for (int r = 0; r < 8; ++r) {
            const int s = q0 + rb + r;
            Ob[(size_t)s * 2048 + f] = f2us(oacc[vt][r] / lrow[r]);
        }
    }
}

// ---------------------------------------------------------------------------
// Host side
// ---------------------------------------------------------------------------
extern "C" void kernel_launch(void* const* d_in, const int* in_sizes, int n_in,
                              void* d_out, int out_size, void* d_ws, size_t ws_size,
                              hipStream_t stream) {
    (void)in_sizes; (void)n_in; (void)out_size; (void)ws_size;
    const float* x        = (const float*)d_in[0];
    // d_in[1] = mask (bool) -- reference mask is plain causal; handled analytically.
    const float* q_kernel = (const float*)d_in[2];  // [C][2048]  (m, rhk)
    const float* k_kernel = (const float*)d_in[3];  // [C][512]
    const float* v_kernel = (const float*)d_in[4];  // [C][512]
    const float* o_kernel = (const float*)d_in[5];  // [2048][C]  (rhv, m)
    float* out = (float*)d_out;

    unsigned short* ws = (unsigned short*)d_ws;
    const size_t SZ_TC = (size_t)T_LEN * C_DIM;     // 4M elements
    const size_t SZ_TK = (size_t)T_LEN * 512;       // 1M elements
    unsigned short* xb  = ws;                 // x bf16            [T][C]
    unsigned short* qwT = xb  + SZ_TC;        // q weights^T       [2048][C]
    unsigned short* kwT = qwT + SZ_TC;        // k weights^T       [512][C]
    unsigned short* vwT = kwT + SZ_TK;        // v weights^T       [512][C]
    unsigned short* woT = vwT + SZ_TK;        // out weights^T     [C][2048]
    unsigned short* Qb  = woT + SZ_TC;        // Q (post-RoPE)     [T][2048]
    unsigned short* Kb  = Qb  + SZ_TC;        // K (post-RoPE)     [T][512]
    unsigned short* Vb  = Kb  + SZ_TK;        // V                 [T][512]
    unsigned short* VT  = Vb  + SZ_TK;        // V^T               [512][T]
    unsigned short* Ob  = VT  + SZ_TK;        // attention output  [T][2048]

    const int TPB = 256;
    const int nTC = (int)SZ_TC;

    // 1. Precision conversion + weight transposes
    cvt_f32_bf16<<<(nTC + TPB - 1) / TPB, TPB, 0, stream>>>(x, xb, nTC);
    transpose_cvt_f32_bf16<<<(nTC + TPB - 1) / TPB, TPB, 0, stream>>>(q_kernel, qwT, C_DIM, 2048);
    transpose_cvt_f32_bf16<<<((int)SZ_TK + TPB - 1) / TPB, TPB, 0, stream>>>(k_kernel, kwT, C_DIM, 512);
    transpose_cvt_f32_bf16<<<((int)SZ_TK + TPB - 1) / TPB, TPB, 0, stream>>>(v_kernel, vwT, C_DIM, 512);
    transpose_cvt_f32_bf16<<<(nTC + TPB - 1) / TPB, TPB, 0, stream>>>(o_kernel, woT, 2048, C_DIM);

    // 2. QKV projections (WMMA GEMMs, async-LDS staged B tiles)
    gemm_bf16_kernel<0><<<dim3(2048 / 64, T_LEN / 128), TPB, 0, stream>>>(xb, qwT, Qb, T_LEN, 2048, C_DIM);
    gemm_bf16_kernel<0><<<dim3(512 / 64, T_LEN / 128), TPB, 0, stream>>>(xb, kwT, Kb, T_LEN, 512, C_DIM);
    gemm_bf16_kernel<0><<<dim3(512 / 64, T_LEN / 128), TPB, 0, stream>>>(xb, vwT, Vb, T_LEN, 512, C_DIM);

    // 3. RoPE (in place)
    rope_kernel<<<(T_LEN * N_HEADS * 64 + TPB - 1) / TPB, TPB, 0, stream>>>(Qb, N_HEADS, 2048);
    rope_kernel<<<(T_LEN * N_KV * 64 + TPB - 1) / TPB, TPB, 0, stream>>>(Kb, N_KV, 512);

    // 4. V transpose for contiguous PV B-fragments
    transpose_bf16<<<((int)SZ_TK + TPB - 1) / TPB, TPB, 0, stream>>>(Vb, VT, T_LEN, 512);

    // 5. Flash attention (softcap + causal + online softmax)
    flash_attn_kernel<<<dim3(T_LEN / 128, N_HEADS), TPB, 0, stream>>>(Qb, Kb, VT, Ob);

    // 6. Output projection (f32 store)
    gemm_bf16_kernel<1><<<dim3(2048 / 64, T_LEN / 128), TPB, 0, stream>>>(Ob, woT, out, T_LEN, C_DIM, 2048);
}
